// FA_21182778704944
// MI455X (gfx1250) — matrix operand order repaired and checked by reading, not compiled, not run on hardware
//
#include <hip/hip_runtime.h>
#include <hip/hip_bf16.h>

// ---------- types ----------
typedef _Float16 v16h __attribute__((ext_vector_type(16)));
typedef _Float16 v8h  __attribute__((ext_vector_type(8)));
typedef float    v8f  __attribute__((ext_vector_type(8)));

union V16 { v16h v; v8h h[2]; };

// ---------- geometry ----------
#define DIMC 224
#define NTOK 64
#define NH   8
#define DH   28

// LDS strides (halves / floats). All chosen so stride/4 (dwords) = 4*odd
// => 16B aligned AND bank-conflict free across 16 lanes.
#define XSTR 232   // x / o rows (224 data halves)
#define QSTR 264   // q / k rows (8 heads * 32 padded halves)
#define VSTR 72    // vT token stride (64 data halves)
#define VHST (28*VSTR)
#define SSTR 68    // sim f32 row stride (64 data floats)
#define ASTR 72    // attn f16 row stride

#define SMEM_BYTES 159744

// weight pack offsets (halves) inside d_ws
#define WQ_OFF 0
#define WV_OFF 100352   // 448*224
#define WO_OFF 150528   // + 224*224

// ---------- fragment loaders ----------
// A fragment (16x32 f16): row-major source, lane l -> row m0+(l&15),
// halves {kb+koff .. +7} and {kb+koff+16 .. +23}, koff = (l&16)?8:0.
__device__ inline v16h load_a(const _Float16* __restrict__ base, int stride,
                              int m0, int kb, int lane) {
  const _Float16* p = base + (m0 + (lane & 15)) * stride + kb + ((lane & 16) ? 8 : 0);
  V16 r;
  r.h[0] = *(const v8h*)p;
  r.h[1] = *(const v8h*)(p + 16);
  return r.v;
}

// B fragment (32x16 f16) from Bt[n][k] (column-contiguous-in-k) storage:
// lane l -> col n0+(l&15), halves {kb+koff .. +7} and {kb+koff+16 .. +23}, koff = (l&16)?8:0.
__device__ inline v16h load_bt(const _Float16* __restrict__ bt, int stride,
                               int n0, int kb, int lane) {
  const _Float16* p = bt + (n0 + (lane & 15)) * stride + kb + ((lane & 16) ? 8 : 0);
  V16 r;
  r.h[0] = *(const v8h*)p;
  r.h[1] = *(const v8h*)(p + 16);
  return r.v;
}

__device__ inline v8f wmma16(v16h a, v16h b, v8f c) {
  return __builtin_amdgcn_wmma_f32_16x16x32_f16(false, a, false, b, (short)0, c,
                                                false, false);
}

// ---------- weight prepack: f32 (in,out) -> f16 Bt[out][in] ----------
__global__ void __launch_bounds__(256)
prepack_kernel(const float* __restrict__ wqk, const float* __restrict__ wv,
               const float* __restrict__ wo, _Float16* __restrict__ wT) {
  int i = blockIdx.x * 256 + threadIdx.x;
  if (i < 100352) {                       // w_qk: [224][448] -> Bt[448][224]
    int n = i / 224, k = i - n * 224;
    wT[WQ_OFF + i] = (_Float16)wqk[k * 448 + n];
  } else if (i < 150528) {                // w_v: [224][224]
    int j = i - 100352;
    int n = j / 224, k = j - n * 224;
    wT[i] = (_Float16)wv[k * 224 + n];
  } else if (i < 200704) {                // w_out: [224][224]
    int j = i - 150528;
    int n = j / 224, k = j - n * 224;
    wT[i] = (_Float16)wo[k * 224 + n];
  }
}

// ---------- main kernel: one window per block ----------
__global__ void __launch_bounds__(256, 1)
fa_window_kernel(const float* __restrict__ x, const _Float16* __restrict__ wT,
                 const float* __restrict__ b_out, const float* __restrict__ pcq_w,
                 const float* __restrict__ pcq_b, const float* __restrict__ pck_w,
                 const float* __restrict__ pck_b, const float* __restrict__ mlp1_w,
                 const float* __restrict__ mlp2_w1, const float* __restrict__ mlp2_w2,
                 float* __restrict__ out) {
  extern __shared__ char smem[];
  _Float16* xo  = (_Float16*)(smem);            // 64*XSTR halves (x, later o)
  _Float16* qs  = (_Float16*)(smem + 29696);    // 64*QSTR
  _Float16* ks  = (_Float16*)(smem + 63488);    // 64*QSTR
  _Float16* vt  = (_Float16*)(smem + 97280);    // 8*28*VSTR (vT[h][d][tok])
  float*    sim = (float*)   (smem + 129536);   // 64*SSTR
  _Float16* at  = (_Float16*)(smem + 146944);   // 64*ASTR
  float*    t1  = (float*)   (smem + 156160);   // 64
  float*    Sqv = (float*)   (smem + 156416);   // 64
  float*    Skv = (float*)   (smem + 156672);   // 64
  float*    h1w = (float*)   (smem + 156928);   // 64
  float*    redA= (float*)   (smem + 157184);   // 4*64
  float*    redB= (float*)   (smem + 158208);   // 4*64
  float*    thv = (float*)   (smem + 159232);   // 1

  const int tid  = threadIdx.x;
  const int wave = tid >> 5, lane = tid & 31;
  const int l16  = lane & 15;
  const int hi8  = (lane & 16) ? 8 : 0;

  const int bid  = blockIdx.x;
  const int bimg = bid >> 10;
  const int wy   = (bid >> 5) & 31;
  const int wx   = bid & 31;

  const _Float16* wqT = wT + WQ_OFF;
  const _Float16* woT = wT + WO_OFF;

  // ---- Phase A: gather shifted window, f32 -> f16; zero q/k K-pad ----
  for (int idx = tid; idx < NTOK * DIMC; idx += 256) {
    int t = idx / DIMC, ch = idx - t * DIMC;
    int r = t >> 3, cc = t & 7;
    int gi = (wy * 8 + r + 252) & 255;     // roll(+4): src = (coord-4) mod 256
    int gj = (wx * 8 + cc + 252) & 255;
    xo[t * XSTR + ch] =
        (_Float16)x[(((size_t)bimg * 256 + gi) * 256 + gj) * DIMC + ch];
  }
  for (int i = tid; i < 2048; i += 256) {  // zero d=28..31 per head in q,k
    int t = i >> 5, rest = i & 31;
    int h = rest >> 2, p = 28 + (rest & 3);
    qs[t * QSTR + h * 32 + p] = (_Float16)0.0f;
    ks[t * QSTR + h * 32 + p] = (_Float16)0.0f;
  }
  __syncthreads();

  // ---- Phase B: QKV projections (168 16x16 tiles, K=224) ----
  for (int t = wave; t < 168; t += 8) {
    int mt = t / 42, nt = t - mt * 42;
    const _Float16* bt;
    int ncol;
    if (nt < 28) { bt = wqT; ncol = nt * 16; }            // qk halves
    else         { bt = wT + WV_OFF; ncol = (nt - 28) * 16; }  // v
    v8f acc = {0.f, 0.f, 0.f, 0.f, 0.f, 0.f, 0.f, 0.f};
    #pragma unroll
    for (int kt = 0; kt < 7; ++kt) {
      v16h a = load_a(xo, XSTR, mt * 16, kt * 32, lane);
      v16h b = load_bt(bt, 224, ncol, kt * 32, lane);
      acc = wmma16(a, b, acc);
    }
    int nglob = ncol + l16;
    #pragma unroll
    for (int g = 0; g < 8; ++g) {
      int m = mt * 16 + g + hi8;
      _Float16 v = (_Float16)acc[g];
      if (nt < 14) { int h = nglob / 28, d = nglob - h * 28; qs[m * QSTR + h * 32 + d] = v; }
      else if (nt < 28) { int c = nglob - 224; int h = c / 28, d = c - h * 28;
                          ks[m * QSTR + h * 32 + d] = v; }
      else { int h = nglob / 28, d = nglob - h * 28; vt[h * VHST + d * VSTR + m] = v; }
    }
  }
  __syncthreads();

  // ---- per-head attention ----
  for (int h = 0; h < NH; ++h) {
    // C1: sim = q @ k^T (k row-major *is* Bt layout), K=32 (padded 28)
    for (int t = wave; t < 16; t += 8) {
      int mt = t >> 2, nt = t & 3;
      v16h a = load_a(qs, QSTR, mt * 16, h * 32, lane);
      v16h b = load_bt(ks, QSTR, nt * 16, h * 32, lane);
      v8f acc = {0.f, 0.f, 0.f, 0.f, 0.f, 0.f, 0.f, 0.f};
      acc = wmma16(a, b, acc);
      #pragma unroll
      for (int g = 0; g < 8; ++g)
        sim[(mt * 16 + g + hi8) * SSTR + nt * 16 + l16] = acc[g];
    }
    __syncthreads();

    // C2a: t1 (raw sim, diag removed) + Sq / Sk sigma scalars
    if (tid < 64) {
      int i = tid;
      float acc = 0.f;
      for (int j = 0; j < 64; ++j) acc += sim[i * SSTR + j] * mlp1_w[j];
      acc -= sim[i * SSTR + i] * mlp1_w[i];
      t1[i] = acc;
      float s = pcq_b[0];
      for (int d = 0; d < DH; ++d) s += (float)qs[i * QSTR + h * 32 + d] * pcq_w[d];
      Sqv[i] = s;
    } else if (tid < 128) {
      int i = tid - 64;
      float s = pck_b[0];
      for (int d = 0; d < DH; ++d) s += (float)ks[i * QSTR + h * 32 + d] * pck_w[d];
      Skv[i] = s;
    }
    __syncthreads();

    // C2b: h1 = t1 @ mlp2_w1, LeakyReLU(0.1), premultiply by mlp2_w2
    if (tid < 64) {
      int j = tid;
      float acc = 0.f;
      for (int i = 0; i < 64; ++i) acc += t1[i] * mlp2_w1[i * 64 + j];
      acc = (acc >= 0.f) ? acc : 0.1f * acc;
      h1w[j] = acc * mlp2_w2[j];
    }
    __syncthreads();
    if (tid == 0) {
      float th = 0.f;
      for (int j = 0; j < 64; ++j) th += h1w[j];
      thv[0] = th;
    }
    __syncthreads();

    // C3: sigma modulation + masked softmax (4 threads/row x 16 cols)
    {
      int part = tid >> 6, row = tid & 63, c0 = part * 16;
      float sq = Sqv[row];
      float pmax = -3.4e38f;
      #pragma unroll
      for (int j = 0; j < 16; ++j) {
        int c = c0 + j;
        float s = sim[row * SSTR + c] * sq * Skv[c];
        sim[row * SSTR + c] = s;
        pmax = fmaxf(pmax, s);
      }
      redA[part * 64 + row] = pmax;
      __syncthreads();
      float rmax = fmaxf(fmaxf(redA[row], redA[64 + row]),
                         fmaxf(redA[128 + row], redA[192 + row]));
      float th = thv[0];
      float psum = 0.f;
      #pragma unroll
      for (int j = 0; j < 16; ++j) {
        int c = c0 + j;
        float s = sim[row * SSTR + c];
        float e = __expf(s - rmax);
        psum += e;
        sim[row * SSTR + c] = (s > th) ? e : -e;  // sign carries the mask
      }
      redB[part * 64 + row] = psum;
      __syncthreads();
      float tot = redB[row] + redB[64 + row] + redB[128 + row] + redB[192 + row];
      float inv = 1.0f / tot;
      #pragma unroll
      for (int j = 0; j < 16; ++j) {
        int c = c0 + j;
        float e = sim[row * SSTR + c];
        at[row * ASTR + c] = (_Float16)((e > 0.f) ? e * inv : 0.f);
      }
    }
    __syncthreads();

    // C4: o_h = attn @ v (A=attn f16, B=vT; K=64), 8 tiles = 1 per wave
    {
      int mt = wave >> 1, nt = wave & 1;
      v8f acc = {0.f, 0.f, 0.f, 0.f, 0.f, 0.f, 0.f, 0.f};
      #pragma unroll
      for (int kt = 0; kt < 2; ++kt) {
        v16h a = load_a(at, ASTR, mt * 16, kt * 32, lane);
        v16h b = load_bt(vt + h * VHST, VSTR, nt * 16, kt * 32, lane);
        acc = wmma16(a, b, acc);
      }
      int d = nt * 16 + l16;
      if (d < DH) {
        #pragma unroll
        for (int g = 0; g < 8; ++g) {
          int m = mt * 16 + g + hi8;
          xo[m * XSTR + h * DH + d] = (_Float16)acc[g];  // reuse x region as o
        }
      }
    }
    __syncthreads();
  }

  // ---- Phase D: out = o @ w_out + b_out, scatter with un-shift ----
  for (int t = wave; t < 56; t += 8) {
    int mt = t / 14, nt = t - mt * 14;
    int ncol = nt * 16;
    v8f acc = {0.f, 0.f, 0.f, 0.f, 0.f, 0.f, 0.f, 0.f};
    #pragma unroll
    for (int kt = 0; kt < 7; ++kt) {
      v16h a = load_a(xo, XSTR, mt * 16, kt * 32, lane);
      v16h b = load_bt(woT, 224, ncol, kt * 32, lane);
      acc = wmma16(a, b, acc);
    }
    int n = ncol + l16;
    float bias = b_out[n];
    #pragma unroll
    for (int g = 0; g < 8; ++g) {
      int m = mt * 16 + g + hi8;
      int r = m >> 3, cc = m & 7;
      int gi = (wy * 8 + r + 252) & 255;   // same (coord-4) mapping un-rolls
      int gj = (wx * 8 + cc + 252) & 255;
      out[(((size_t)bimg * 256 + gi) * 256 + gj) * DIMC + n] = acc[g] + bias;
    }
  }
}

// ---------- launcher ----------
extern "C" void kernel_launch(void* const* d_in, const int* in_sizes, int n_in,
                              void* d_out, int out_size, void* d_ws, size_t ws_size,
                              hipStream_t stream) {
  (void)in_sizes; (void)n_in; (void)out_size; (void)ws_size;
  const float* x       = (const float*)d_in[0];
  const float* w_qk    = (const float*)d_in[1];
  const float* w_v     = (const float*)d_in[2];
  const float* w_out   = (const float*)d_in[3];
  const float* b_out   = (const float*)d_in[4];
  const float* pcq_w   = (const float*)d_in[5];
  const float* pcq_b   = (const float*)d_in[6];
  const float* pck_w   = (const float*)d_in[7];
  const float* pck_b   = (const float*)d_in[8];
  const float* mlp1_w  = (const float*)d_in[9];
  const float* mlp2_w1 = (const float*)d_in[10];
  const float* mlp2_w2 = (const float*)d_in[11];
  float* out = (float*)d_out;
  _Float16* wT = (_Float16*)d_ws;

  prepack_kernel<<<(200704 + 255) / 256, 256, 0, stream>>>(w_qk, w_v, w_out, wT);

  hipFuncSetAttribute(reinterpret_cast<const void*>(fa_window_kernel),
                      hipFuncAttributeMaxDynamicSharedMemorySize, SMEM_BYTES);
  fa_window_kernel<<<4096, 256, SMEM_BYTES, stream>>>(
      x, wT, b_out, pcq_w, pcq_b, pck_w, pck_b, mlp1_w, mlp2_w1, mlp2_w2, out);
}
